// MultiHeadAttention_52673478918912
// MI455X (gfx1250) — compile-verified, matrix operand order checked by reading, not run on hardware
//
#include <hip/hip_runtime.h>
#include <hip/hip_bf16.h>

// ---------------------------------------------------------------------------
// Types for CDNA5 WMMA (wave32): v_wmma_f32_16x16x32_bf16
// ---------------------------------------------------------------------------
typedef __bf16 bf16_t;
typedef __attribute__((ext_vector_type(16))) __bf16 v16bf;
typedef __attribute__((ext_vector_type(8)))  __bf16 v8bf;
typedef __attribute__((ext_vector_type(4)))  __bf16 v4bf;
typedef __attribute__((ext_vector_type(8)))  float  v8f;
typedef __attribute__((ext_vector_type(4)))  int    v4i;

#define S_LEN   2048
#define DM      512
#define NHEADS  8
#define DEPTH   64
#define BATCH   4

#if __has_builtin(__builtin_amdgcn_global_load_async_to_lds_b128) && \
    __has_builtin(__builtin_amdgcn_s_wait_asynccnt)
#define USE_ASYNC_LDS 1
#else
#define USE_ASYNC_LDS 0
#endif

__device__ __forceinline__ v8f wmma_bf16(v16bf a, v16bf b, v8f c) {
    return __builtin_amdgcn_wmma_f32_16x16x32_bf16(
        false, a, false, b, (short)0, c, false, false);
}

// Row-major fragment load (A-frag, or B-frag from a tile stored transposed).
// ISA 16-bit 16x32 layout per lane:
//   lanes 0-15 : row = lane%16, elems 0..7 -> K=k0+0..7,  8..15 -> K=k0+16..23
//   lanes16-31 : row = lane%16, elems 0..7 -> K=k0+8..15, 8..15 -> K=k0+24..31
__device__ __forceinline__ v16bf frag_rowmajor(const bf16_t* tile, int ld,
                                               int r, int k0, int hf) {
    const bf16_t* p = tile + (size_t)r * ld + k0 + hf * 8;
    v8bf lo = *(const v8bf*)(p);
    v8bf hi = *(const v8bf*)(p + 16);
    return __builtin_shufflevector(lo, hi, 0, 1, 2, 3, 4, 5, 6, 7,
                                           8, 9, 10, 11, 12, 13, 14, 15);
}

// f32x4 -> bf16x4 (lowers to 2x v_cvt_pk_bf16_f32 + b64 store)
__device__ __forceinline__ v4bf cvt4bf(float4 f) {
    v4bf t;
    t[0] = (bf16_t)f.x; t[1] = (bf16_t)f.y;
    t[2] = (bf16_t)f.z; t[3] = (bf16_t)f.w;
    return t;
}

// 16-byte global -> LDS copy.  Async DMA (ASYNCcnt) when available.
__device__ __forceinline__ void copy16_to_lds(bf16_t* l, const bf16_t* g) {
#if USE_ASYNC_LDS
    __builtin_amdgcn_global_load_async_to_lds_b128(
        (__attribute__((address_space(1))) v4i*)g,
        (__attribute__((address_space(3))) v4i*)l, 0, 0);
#else
    *(v8bf*)l = *(const v8bf*)g;
#endif
}
__device__ __forceinline__ void wait_async_lds() {
#if USE_ASYNC_LDS
    __builtin_amdgcn_s_wait_asynccnt(0);
#endif
}

// ---------------------------------------------------------------------------
// GEMM  C[M,N] = A[M,K](f32) * W[K,N](f32) + bias, bf16 WMMA, f32 accum.
// Double-buffered LDS, register prefetch, W tile staged transposed.
// mode 0: write bf16 split-head layout [B,H,S,64]; mode 1: f32 row-major.
// ---------------------------------------------------------------------------
#define GTM 128
#define GTN 64
#define GTK 32
#define GLDA (GTK + 8)
#define GLDB (GTK + 8)

__global__ __launch_bounds__(256) void gemm_bias_kernel(
    const float* __restrict__ A, const float* __restrict__ W,
    const float* __restrict__ bias, bf16_t* __restrict__ out_bf,
    float* __restrict__ out_f, int M, int N, int K, int mode)
{
    __shared__ bf16_t lA [2][GTM][GLDA];
    __shared__ bf16_t lBt[2][GTN][GLDB];   // W transposed: [n][k]

    const int tid  = threadIdx.x;
    const int wave = tid >> 5;
    const int lane = tid & 31;
    const int hf   = lane >> 4;
    const int lr   = lane & 15;
    const int m0   = blockIdx.y * GTM;
    const int n0   = blockIdx.x * GTN;

    const int rA = tid >> 3;            // + it*32 (it = 0..3)
    const int cA = (tid & 7) * 4;
    const int rW = tid >> 4;            // + it*16 (it = 0..1)
    const int cW = (tid & 15) * 4;

    const float* Abase = A + (size_t)m0 * K;

    float4 pa[4], pw[2];
#pragma unroll
    for (int it = 0; it < 4; ++it)
        pa[it] = *(const float4*)(Abase + (size_t)(rA + it * 32) * K + cA);
#pragma unroll
    for (int it = 0; it < 2; ++it)
        pw[it] = *(const float4*)(W + (size_t)(rW + it * 16) * N + n0 + cW);

    v8f acc[4];
#pragma unroll
    for (int nb = 0; nb < 4; ++nb) acc[nb] = (v8f){0,0,0,0,0,0,0,0};

    const int T = K / GTK;   // 16
    for (int t = 0; t < T; ++t) {
        const int cur = t & 1;
#pragma unroll
        for (int it = 0; it < 4; ++it)
            *(v4bf*)&lA[cur][rA + it * 32][cA] = cvt4bf(pa[it]);
#pragma unroll
        for (int it = 0; it < 2; ++it) {
            int r = rW + it * 16;
            lBt[cur][cW + 0][r] = (bf16_t)pw[it].x;
            lBt[cur][cW + 1][r] = (bf16_t)pw[it].y;
            lBt[cur][cW + 2][r] = (bf16_t)pw[it].z;
            lBt[cur][cW + 3][r] = (bf16_t)pw[it].w;
        }
        __syncthreads();

        if (t + 1 < T) {
            const int kb = (t + 1) * GTK;
#pragma unroll
            for (int it = 0; it < 4; ++it)
                pa[it] = *(const float4*)(Abase + (size_t)(rA + it * 32) * K + kb + cA);
#pragma unroll
            for (int it = 0; it < 2; ++it)
                pw[it] = *(const float4*)(W + (size_t)(kb + rW + it * 16) * N + n0 + cW);
        }

        v16bf a = frag_rowmajor(&lA[cur][wave * 16][0], GLDA, lr, 0, hf);
        v16bf b[4];
#pragma unroll
        for (int nb = 0; nb < 4; ++nb)
            b[nb] = frag_rowmajor(&lBt[cur][nb * 16][0], GLDB, lr, 0, hf);
#pragma unroll
        for (int nb = 0; nb < 4; ++nb)
            acc[nb] = wmma_bf16(a, b[nb], acc[nb]);
        __syncthreads();
    }

#pragma unroll
    for (int nb = 0; nb < 4; ++nb) {
#pragma unroll
        for (int j = 0; j < 8; ++j) {
            int row = m0 + wave * 16 + j + hf * 8;
            int col = n0 + nb * 16 + lr;
            float val = acc[nb][j] + bias[col];
            if (mode == 0) {
                int b2 = row >> 11, s = row & (S_LEN - 1);
                int h  = col >> 6,  d = col & (DEPTH - 1);
                out_bf[(((size_t)(b2 * NHEADS + h) * S_LEN) + s) * DEPTH + d] =
                    (bf16_t)val;
            } else {
                out_f[(size_t)row * N + col] = val;
            }
        }
    }
}

// ---------------------------------------------------------------------------
// Flash attention. grid = (S/128, B*H), 256 threads = 8 waves.
// K tiles: async global->LDS DMA prefetch (double-buffered).
// V tiles: register prefetch + transposed LDS staging ([d][key]).
// ctx layout: [B*S, 512] f32 (heads concatenated).
// ---------------------------------------------------------------------------
#define FBQ 128
#define FBK 64
#define FLDK (DEPTH + 8)
#define FLDV (FBK + 8)
#define FLDP (DEPTH + 8)

__global__ __launch_bounds__(256) void flash_attn_kernel(
    const bf16_t* __restrict__ Qh, const bf16_t* __restrict__ Kh,
    const bf16_t* __restrict__ Vh, const float* __restrict__ mask,
    float* __restrict__ ctx)
{
    __shared__ bf16_t lK [2][FBK][FLDK];    // keys x depth
    __shared__ bf16_t lVt[2][DEPTH][FLDV];  // depth x keys (transposed)
    __shared__ bf16_t lP [8][16][FLDP];     // per-wave P tile

    const int tid  = threadIdx.x;
    const int wave = tid >> 5;
    const int lane = tid & 31;
    const int hf   = lane >> 4;
    const int lr   = lane & 15;

    const int bh = blockIdx.y;
    const int q0 = blockIdx.x * FBQ;
    const bf16_t* Qb = Qh + (size_t)bh * S_LEN * DEPTH;
    const bf16_t* Kb = Kh + (size_t)bh * S_LEN * DEPTH;
    const bf16_t* Vb = Vh + (size_t)bh * S_LEN * DEPTH;

    const int rS = tid >> 3;            // + it*32 (it = 0..1)
    const int cS = (tid & 7) * 8;

    v16bf qa0 = frag_rowmajor(Qb + (size_t)(q0 + wave * 16) * DEPTH, DEPTH, lr, 0,  hf);
    v16bf qa1 = frag_rowmajor(Qb + (size_t)(q0 + wave * 16) * DEPTH, DEPTH, lr, 32, hf);

    float mrow[8], lsum[8];
    v8f o[4];
#pragma unroll
    for (int j = 0; j < 8; ++j) { mrow[j] = -3.0e38f; lsum[j] = 0.0f; }
#pragma unroll
    for (int nb = 0; nb < 4; ++nb) o[nb] = (v8f){0,0,0,0,0,0,0,0};

    const float scale = 0.125f;
    const int T = S_LEN / FBK;    // 32

    // prologue: async-copy K tile 0 into lK[0]; prefetch V tile 0 into regs
    v8bf pv[2];
#pragma unroll
    for (int it = 0; it < 2; ++it) {
        int r = rS + it * 32;
        copy16_to_lds(&lK[0][r][cS], Kb + (size_t)r * DEPTH + cS);
        pv[it] = *(const v8bf*)(Vb + (size_t)r * DEPTH + cS);
    }

    for (int t = 0; t < T; ++t) {
        const int cur = t & 1, nxt = cur ^ 1;

        // stage V tile t (transposed) from regs into lVt[cur]
#pragma unroll
        for (int it = 0; it < 2; ++it) {
            int r = rS + it * 32;
#pragma unroll
            for (int i = 0; i < 8; ++i) lVt[cur][cS + i][r] = pv[it][i];
        }
        wait_async_lds();          // K tile t has landed in lK[cur]
        __syncthreads();

        // prefetch tile t+1 (K: async DMA into lK[nxt]; V: regs).  Safe: all
        // reads of lK[nxt] happened in iteration t-1, before this barrier.
        if (t + 1 < T) {
            const int kb = (t + 1) * FBK;
#pragma unroll
            for (int it = 0; it < 2; ++it) {
                int r = rS + it * 32;
                copy16_to_lds(&lK[nxt][r][cS], Kb + (size_t)(kb + r) * DEPTH + cS);
                pv[it] = *(const v8bf*)(Vb + (size_t)(kb + r) * DEPTH + cS);
            }
        }

        // ----- logits S = Q * K^T  (16 x 64 per wave), batched frags
        v8f sacc[4];
        {
            v16bf bk[4];
#pragma unroll
            for (int nb = 0; nb < 4; ++nb)
                bk[nb] = frag_rowmajor(&lK[cur][nb * 16][0], FLDK, lr, 0, hf);
#pragma unroll
            for (int nb = 0; nb < 4; ++nb) {
                sacc[nb] = (v8f){0,0,0,0,0,0,0,0};
                sacc[nb] = wmma_bf16(qa0, bk[nb], sacc[nb]);
            }
#pragma unroll
            for (int nb = 0; nb < 4; ++nb)
                bk[nb] = frag_rowmajor(&lK[cur][nb * 16][0], FLDK, lr, 32, hf);
#pragma unroll
            for (int nb = 0; nb < 4; ++nb)
                sacc[nb] = wmma_bf16(qa1, bk[nb], sacc[nb]);
        }

        // ----- online softmax
        const int kb0 = t * FBK;
#pragma unroll
        for (int j = 0; j < 8; ++j) {
            int qrow = q0 + wave * 16 + j + hf * 8;
            float vmax = -3.0e38f;
#pragma unroll
            for (int nb = 0; nb < 4; ++nb) {
                int key = kb0 + nb * 16 + lr;
                float s = sacc[nb][j] * scale +
                          mask[(size_t)qrow * S_LEN + key] * (-1.0e9f);
                sacc[nb][j] = s;
                vmax = fmaxf(vmax, s);
            }
            for (int m = 1; m < 16; m <<= 1)
                vmax = fmaxf(vmax, __shfl_xor(vmax, m, 32));
            float mnew  = fmaxf(mrow[j], vmax);
            float alpha = __expf(mrow[j] - mnew);
            float rsum  = 0.0f;
#pragma unroll
            for (int nb = 0; nb < 4; ++nb) {
                float p = __expf(sacc[nb][j] - mnew);
                sacc[nb][j] = p;
                rsum += p;
            }
            for (int m = 1; m < 16; m <<= 1)
                rsum += __shfl_xor(rsum, m, 32);
            lsum[j] = lsum[j] * alpha + rsum;
            mrow[j] = mnew;
#pragma unroll
            for (int nb = 0; nb < 4; ++nb) o[nb][j] *= alpha;
        }

        // ----- re-stripe P (C-layout -> row-major bf16 in LDS)
#pragma unroll
        for (int j = 0; j < 8; ++j) {
            int pr = j + hf * 8;
#pragma unroll
            for (int nb = 0; nb < 4; ++nb)
                lP[wave][pr][nb * 16 + lr] = (bf16_t)sacc[nb][j];
        }
        __syncthreads();

        // ----- O += P * V (batched frags)
#pragma unroll
        for (int kk = 0; kk < 2; ++kk) {
            v16bf pa = frag_rowmajor(&lP[wave][0][0], FLDP, lr, kk * 32, hf);
            v16bf bv[4];
#pragma unroll
            for (int nb = 0; nb < 4; ++nb)
                bv[nb] = frag_rowmajor(&lVt[cur][nb * 16][0], FLDV, lr, kk * 32, hf);
#pragma unroll
            for (int nb = 0; nb < 4; ++nb)
                o[nb] = wmma_bf16(pa, bv[nb], o[nb]);
        }
    }

    // epilogue
    const int h = bh & (NHEADS - 1);
    const int b = bh >> 3;
#pragma unroll
    for (int nb = 0; nb < 4; ++nb) {
#pragma unroll
        for (int j = 0; j < 8; ++j) {
            int srow = q0 + wave * 16 + j + hf * 8;
            int d    = nb * 16 + lr;
            ctx[((size_t)(b * S_LEN + srow)) * DM + h * DEPTH + d] =
                o[nb][j] / lsum[j];
        }
    }
}

// ---------------------------------------------------------------------------
// Host-side launcher
// ---------------------------------------------------------------------------
extern "C" void kernel_launch(void* const* d_in, const int* in_sizes, int n_in,
                              void* d_out, int out_size, void* d_ws, size_t ws_size,
                              hipStream_t stream) {
    const float* q    = (const float*)d_in[0];
    const float* k    = (const float*)d_in[1];
    const float* v    = (const float*)d_in[2];
    const float* mask = (const float*)d_in[3];
    const float* wq   = (const float*)d_in[4];
    const float* bq   = (const float*)d_in[5];
    const float* wk   = (const float*)d_in[6];
    const float* bk   = (const float*)d_in[7];
    const float* wv   = (const float*)d_in[8];
    const float* bv   = (const float*)d_in[9];
    const float* wo   = (const float*)d_in[10];
    const float* bo   = (const float*)d_in[11];

    const int M = BATCH * S_LEN;   // 8192
    const int N = DM;              // 512
    const int K = DM;              // 512

    const size_t head_elems = (size_t)BATCH * NHEADS * S_LEN * DEPTH; // 4M
    bf16_t* qh  = (bf16_t*)d_ws;
    bf16_t* kh  = qh + head_elems;
    bf16_t* vh  = kh + head_elems;
    float*  ctx = (float*)(vh + head_elems);

    dim3 gemm_grid(N / GTN, M / GTM);   // (8, 64)
    dim3 gemm_block(256);

    gemm_bias_kernel<<<gemm_grid, gemm_block, 0, stream>>>(
        q, wq, bq, qh, nullptr, M, N, K, 0);
    gemm_bias_kernel<<<gemm_grid, gemm_block, 0, stream>>>(
        k, wk, bk, kh, nullptr, M, N, K, 0);
    gemm_bias_kernel<<<gemm_grid, gemm_block, 0, stream>>>(
        v, wv, bv, vh, nullptr, M, N, K, 0);

    dim3 fa_grid(S_LEN / FBQ, BATCH * NHEADS);  // (16, 32)
    flash_attn_kernel<<<fa_grid, dim3(256), 0, stream>>>(qh, kh, vh, mask, ctx);

    gemm_bias_kernel<<<gemm_grid, gemm_block, 0, stream>>>(
        ctx, wo, bo, nullptr, (float*)d_out, M, N, K, 1);
}
